// DualPathSoftMoELayerWrapper_3719441678626
// MI455X (gfx1250) — matrix operand order repaired and checked by reading, not compiled, not run on hardware
//
#include <hip/hip_runtime.h>
#include <math.h>

// Problem constants (from reference)
#define B_   4
#define N_   4096
#define DIM_ 768
#define H_   12
#define HD_  64
#define E_   160
#define NC_  32
#define NU_  128
#define CH_  3072
#define OH_  768

typedef __attribute__((ext_vector_type(16))) __bf16 v16bf;
typedef __attribute__((ext_vector_type(8)))  __bf16 v8bf;
typedef __attribute__((ext_vector_type(8)))  float  v8f;

__device__ __forceinline__ v8f wmma_bf16(v16bf a, v16bf b, v8f c) {
  // D = A(16x32 bf16) * B(32x16 bf16) + C(16x16 f32)
  return __builtin_amdgcn_wmma_f32_16x16x32_bf16(
      /*neg_a=*/false, a, /*neg_b=*/false, b,
      /*c_mod=*/(short)0, c, /*reuse_a=*/false, /*reuse_b=*/false);
}

// Build the 16x32 bf16 A operand for one wave (row = lane%16, contiguous
// 16-wide K chunk k = 16*(lane/16) + s, matching the B operand).
__device__ __forceinline__ v16bf load_a(const __bf16* arow, int kt, int hi) {
  v8bf a0 = *(const v8bf*)(arow + kt + hi * 16);
  v8bf a1 = *(const v8bf*)(arow + kt + hi * 16 + 8);
  v16bf a;
#pragma unroll
  for (int i = 0; i < 8; ++i) { a[i] = a0[i]; a[8 + i] = a1[i]; }
  return a;
}

// ---------------------------------------------------------------------------
// K1: q = l2norm(layernorm(phi*qg+qb)) -> bf16 (H,E,HD). One block per (h,e).
// ---------------------------------------------------------------------------
__global__ void prep_q_kernel(const float* __restrict__ phi,
                              const float* __restrict__ qg, const float* __restrict__ qb,
                              const float* __restrict__ lw, const float* __restrict__ lb,
                              __bf16* __restrict__ qn_bf) {
  int he = blockIdx.x;                 // h*E + e
  int h = he / E_, e = he % E_;
  int d = threadIdx.x;                 // 0..63
  __shared__ float red[64];
  __shared__ float sm, sv, sn;
  float t = phi[(size_t)e * DIM_ + h * HD_ + d] * qg[d] + qb[d];
  red[d] = t; __syncthreads();
  for (int off = 32; off > 0; off >>= 1) { if (d < off) red[d] += red[d + off]; __syncthreads(); }
  if (d == 0) sm = red[0] * (1.0f / 64.0f);
  __syncthreads();
  float c = t - sm;
  red[d] = c * c; __syncthreads();
  for (int off = 32; off > 0; off >>= 1) { if (d < off) red[d] += red[d + off]; __syncthreads(); }
  if (d == 0) sv = rsqrtf(red[0] * (1.0f / 64.0f) + 1e-5f);
  __syncthreads();
  float y = c * sv * lw[d] + lb[d];
  red[d] = y * y; __syncthreads();
  for (int off = 32; off > 0; off >>= 1) { if (d < off) red[d] += red[d + off]; __syncthreads(); }
  if (d == 0) sn = 1.0f / (sqrtf(red[0]) + 1e-6f);
  __syncthreads();
  qn_bf[(size_t)he * HD_ + d] = (__bf16)(y * sn);
}

// ---------------------------------------------------------------------------
// K2: k = l2norm(xh*kg+kb) -> bf16 (B,H,N,HD). One block per (b,n).
// ---------------------------------------------------------------------------
__global__ void prep_k_kernel(const float* __restrict__ x,
                              const float* __restrict__ kg, const float* __restrict__ kb,
                              __bf16* __restrict__ k_bf) {
  int bn = blockIdx.x;                 // b*N + n
  int b = bn / N_, n = bn % N_;
  int tid = threadIdx.x;               // 0..767
  int h = tid >> 6, d = tid & 63;
  __shared__ float red[DIM_];
  float xv = x[(size_t)bn * DIM_ + tid];
  float t = xv * kg[d] + kb[d];
  red[tid] = t * t;
  __syncthreads();
  for (int off = 32; off > 0; off >>= 1) {
    if (d < off) red[tid] += red[tid + off];
    __syncthreads();
  }
  float nrm = sqrtf(red[h << 6]);
  k_bf[(((size_t)b * H_ + h) * N_ + n) * HD_ + d] = (__bf16)(t / (nrm + 1e-6f));
}

// ---------------------------------------------------------------------------
// K2b: xT (B,H,HD,N) bf16 = transpose of x heads, so the slots GEMM B-operand
// becomes a contiguous 32B per-lane load. LDS 32x32 tile, padded stride.
// ---------------------------------------------------------------------------
__global__ void transpose_x_kernel(const float* __restrict__ x,
                                   __bf16* __restrict__ xT) {
  __shared__ __bf16 tile[32][33];
  int t = blockIdx.x;
  int dt = t & 1; t >>= 1;             // 2 chunks of 32 dims
  int nt = t % (N_ / 32);
  int bh = t / (N_ / 32);              // b*H + h
  int b = bh / H_, h = bh % H_;
  int c = threadIdx.x & 31;
  int r0 = (threadIdx.x >> 5) * 4;     // 8 waves * 4 rows
#pragma unroll
  for (int rr = 0; rr < 4; ++rr) {
    int n = nt * 32 + r0 + rr;
    tile[r0 + rr][c] = (__bf16)x[((size_t)b * N_ + n) * DIM_ + h * HD_ + dt * 32 + c];
  }
  __syncthreads();
#pragma unroll
  for (int rr = 0; rr < 4; ++rr) {
    int d = dt * 32 + r0 + rr;
    xT[((size_t)bh * HD_ + d) * N_ + nt * 32 + c] = tile[c][r0 + rr];
  }
}

// ---------------------------------------------------------------------------
// K3: logits (B,H,N,E) f32 = K(Nx64) * Qn^T(64xE) per (b,h). One wave per
// 16-token stripe: A kept in registers, swept across all 10 expert tiles.
// ---------------------------------------------------------------------------
__global__ void logits_kernel(const __bf16* __restrict__ k_bf,
                              const __bf16* __restrict__ qn_bf,
                              float* __restrict__ logits) {
  const int lane = threadIdx.x & 31;
  const int wave = blockIdx.x * (blockDim.x >> 5) + (threadIdx.x >> 5);
  int mt = wave % (N_ / 16);
  int bh = wave / (N_ / 16);           // b*H + h
  int h  = bh % H_;
  const int r16 = lane & 15, hi = lane >> 4;
  const __bf16* arow = k_bf + ((size_t)bh * N_ + mt * 16 + r16) * HD_;
  v16bf a0 = load_a(arow, 0, hi);
  v16bf a1 = load_a(arow, 32, hi);
  v8f acc[10];
#pragma unroll
  for (int et = 0; et < 10; ++et) acc[et] = (v8f){};
#pragma unroll
  for (int et = 0; et < 10; ++et) {
    const __bf16* bcol = qn_bf + ((size_t)h * E_ + et * 16 + r16) * HD_;
    v16bf b0 = *(const v16bf*)(bcol + hi * 16);
    v16bf b1 = *(const v16bf*)(bcol + 32 + hi * 16);
    acc[et] = wmma_bf16(a0, b0, acc[et]);
    acc[et] = wmma_bf16(a1, b1, acc[et]);
  }
  float* pC = logits + ((size_t)bh * N_ + mt * 16) * E_;
#pragma unroll
  for (int et = 0; et < 10; ++et)
#pragma unroll
    for (int r = 0; r < 8; ++r)
      pC[(size_t)(hi * 8 + r) * E_ + et * 16 + r16] = acc[et][r];
}

// ---------------------------------------------------------------------------
// K4: dispatch = softmax over tokens (axis n). One block per (b,h); thread e
// walks its column; row reads coalesce across the 160 threads. Output bf16
// (B,H,E,N) = A-matrix of the slots GEMM.
// ---------------------------------------------------------------------------
__global__ void dispatch_kernel(const float* __restrict__ logits,
                                const float* __restrict__ scale0,
                                __bf16* __restrict__ dispT) {
  int bh = blockIdx.x;
  int e  = threadIdx.x;                // 0..159
  float inv = 1.0f / scale0[0];
  const float* p = logits + (size_t)bh * N_ * E_ + e;
  float m = -3.4e38f;
  for (int n = 0; n < N_; ++n) m = fmaxf(m, p[(size_t)n * E_] * inv);
  float s = 0.f;
  for (int n = 0; n < N_; ++n) s += __expf(p[(size_t)n * E_] * inv - m);
  float rs = 1.0f / s;
  __bf16* o = dispT + ((size_t)bh * E_ + e) * N_;
  for (int n = 0; n < N_; ++n)
    o[n] = (__bf16)(__expf(p[(size_t)n * E_] * inv - m) * rs);
}

// ---------------------------------------------------------------------------
// K5: exact sorted-support 1.5-entmax over 160 experts of one token.
// 256-thread block: bitonic sort (desc) + inclusive scans in LDS; support
// count restricted to the 160 real entries.
// ---------------------------------------------------------------------------
__global__ void entmax_kernel(const float* __restrict__ logits,
                              const float* __restrict__ scale1,
                              __bf16* __restrict__ comb) {
  const int token = blockIdx.x;        // bh*N + n
  const int tid = threadIdx.x;
  __shared__ float s[256], c1[256], c2[256], taub[256], red[256];
  float inv = 1.0f / scale1[0];
  float v = (tid < E_) ? logits[(size_t)token * E_ + tid] * inv : -1e30f;
  red[tid] = v; __syncthreads();
  for (int off = 128; off > 0; off >>= 1) { if (tid < off) red[tid] = fmaxf(red[tid], red[tid + off]); __syncthreads(); }
  float vmax = red[0]; __syncthreads();
  float x = (tid < E_) ? (v - vmax) * 0.5f : -1e30f;
  s[tid] = x; __syncthreads();
  for (int k = 2; k <= 256; k <<= 1)
    for (int j = k >> 1; j > 0; j >>= 1) {
      int ixj = tid ^ j;
      if (ixj > tid) {
        float a = s[tid], b = s[ixj];
        bool desc = ((tid & k) == 0);
        if (desc ? (a < b) : (a > b)) { s[tid] = b; s[ixj] = a; }
      }
      __syncthreads();
    }
  float sv = s[tid];
  c1[tid] = sv; c2[tid] = sv * sv; __syncthreads();
  for (int off = 1; off < 256; off <<= 1) {
    float t1 = (tid >= off) ? c1[tid - off] : 0.f;
    float t2 = (tid >= off) ? c2[tid - off] : 0.f;
    __syncthreads();
    c1[tid] += t1; c2[tid] += t2;
    __syncthreads();
  }
  float flag = 0.f;
  if (tid < E_) {
    float rho  = (float)(tid + 1);
    float mean = c1[tid] / rho;
    float msq  = c2[tid] / rho;
    float ss   = rho * (msq - mean * mean);
    float dl   = (1.0f - ss) / rho;
    float tau  = mean - sqrtf(fmaxf(dl, 0.0f));
    taub[tid] = tau;
    flag = (tau <= sv) ? 1.0f : 0.0f;
  } else {
    taub[tid] = 0.f;
  }
  red[tid] = flag; __syncthreads();
  for (int off = 128; off > 0; off >>= 1) { if (tid < off) red[tid] += red[tid + off]; __syncthreads(); }
  int support = (int)(red[0] + 0.5f);
  float tstar = taub[support - 1];
  if (tid < E_) {
    float w = fmaxf(x - tstar, 0.0f);
    comb[(size_t)token * E_ + tid] = (__bf16)(w * w);
  }
}

// ---------------------------------------------------------------------------
// K6: slots (B,E,DIM) f32: per (b,h) C(160x64) = dispatch^T(160xN) * xh(Nx64).
// One wave per (b,h,expert-tile): A registers reused across 4 dim-tiles; B is
// a contiguous per-lane v16bf from xT. K=4096 -> 4x128 WMMA per wave.
// ---------------------------------------------------------------------------
__global__ void slots_kernel(const __bf16* __restrict__ dispT,
                             const __bf16* __restrict__ xT,
                             float* __restrict__ slots) {
  const int lane = threadIdx.x & 31;
  const int wave = blockIdx.x * (blockDim.x >> 5) + (threadIdx.x >> 5);
  int et = wave % 10;
  int bh = wave / 10;
  int b = bh / H_, h = bh % H_;
  const int r16 = lane & 15, hi = lane >> 4;
  const __bf16* arow = dispT + ((size_t)bh * E_ + et * 16 + r16) * N_;
  v8f acc[4];
#pragma unroll
  for (int dt = 0; dt < 4; ++dt) acc[dt] = (v8f){};
  for (int kt = 0; kt < N_; kt += 32) {
    v16bf a = load_a(arow, kt, hi);
#pragma unroll
    for (int dt = 0; dt < 4; ++dt) {
      const __bf16* brow = xT + ((size_t)bh * HD_ + dt * 16 + r16) * N_;
      v16bf bm = *(const v16bf*)(brow + kt + hi * 16);
      acc[dt] = wmma_bf16(a, bm, acc[dt]);
    }
  }
#pragma unroll
  for (int dt = 0; dt < 4; ++dt)
#pragma unroll
    for (int r = 0; r < 8; ++r) {
      int e = et * 16 + hi * 8 + r;
      int d = h * HD_ + dt * 16 + r16;
      slots[((size_t)b * E_ + e) * DIM_ + d] = acc[dt][r];
    }
}

// ---------------------------------------------------------------------------
// K7: expert MLP layer 1 + exact GELU. M=4 GEMV -> pure weight bandwidth.
// Weights streamed once: non-temporal loads keep L2 free for slots reuse.
// ---------------------------------------------------------------------------
__global__ void mlp1_kernel(const float* __restrict__ slots,
                            const float* __restrict__ w1, const float* __restrict__ b1,
                            float* __restrict__ hid,
                            int e_base, int n_exp, int hidden, int nchunk) {
  int e_local = blockIdx.x / nchunk;
  int jc = blockIdx.x % nchunk;
  int tid = threadIdx.x;
  int j = jc * 256 + tid;
  __shared__ float sl[B_][DIM_];
  for (int i = tid; i < B_ * DIM_; i += 256) {
    int b = i / DIM_, d = i % DIM_;
    sl[b][d] = slots[((size_t)b * E_ + e_base + e_local) * DIM_ + d];
  }
  __syncthreads();
  float acc0 = 0, acc1 = 0, acc2 = 0, acc3 = 0;
  const float* wp = w1 + (size_t)e_local * DIM_ * hidden + j;
  for (int d = 0; d < DIM_; ++d) {
    float w = __builtin_nontemporal_load(wp + (size_t)d * hidden);
    acc0 += sl[0][d] * w; acc1 += sl[1][d] * w;
    acc2 += sl[2][d] * w; acc3 += sl[3][d] * w;
  }
  float bb = b1[(size_t)e_local * hidden + j];
  float a[B_] = {acc0 + bb, acc1 + bb, acc2 + bb, acc3 + bb};
#pragma unroll
  for (int b = 0; b < B_; ++b) {
    float g = 0.5f * a[b] * (1.0f + erff(a[b] * 0.70710678118654752f));
    hid[((size_t)b * n_exp + e_local) * hidden + j] = g;
  }
}

// ---------------------------------------------------------------------------
// K8: expert MLP layer 2. Emits eoT (B,DIM,E) bf16 so the combine GEMM
// B-operand is a contiguous per-lane v16bf.
// ---------------------------------------------------------------------------
template <int HID>
__global__ void mlp2_kernel(const float* __restrict__ hid,
                            const float* __restrict__ w2, const float* __restrict__ b2,
                            __bf16* __restrict__ eoT,
                            int e_base, int n_exp) {
  int e_local = blockIdx.x;
  int tid = threadIdx.x;
  __shared__ float hl[B_][HID];
  for (int i = tid; i < B_ * HID; i += 256) {
    int b = i / HID, jj = i % HID;
    hl[b][jj] = hid[((size_t)b * n_exp + e_local) * HID + jj];
  }
  __syncthreads();
  for (int dc = 0; dc < DIM_; dc += 256) {
    int d = dc + tid;
    float acc0 = 0, acc1 = 0, acc2 = 0, acc3 = 0;
    const float* wp = w2 + (size_t)e_local * HID * DIM_ + d;
    for (int jj = 0; jj < HID; ++jj) {
      float w = __builtin_nontemporal_load(wp + (size_t)jj * DIM_);
      acc0 += hl[0][jj] * w; acc1 += hl[1][jj] * w;
      acc2 += hl[2][jj] * w; acc3 += hl[3][jj] * w;
    }
    float bb = b2[(size_t)e_local * DIM_ + d];
    float a[B_] = {acc0 + bb, acc1 + bb, acc2 + bb, acc3 + bb};
#pragma unroll
    for (int b = 0; b < B_; ++b)
      eoT[((size_t)b * DIM_ + d) * E_ + e_base + e_local] = (__bf16)a[b];
  }
}

// ---------------------------------------------------------------------------
// K9: out (B,N,DIM) f32: per (b,h) C(Nx64) = combine(NxE) * eo_h(Ex64).
// One wave per 16-token stripe: A registers reused across 4 dim-tiles; B from
// eoT is contiguous. K=160 -> 20 WMMA per wave.
// ---------------------------------------------------------------------------
__global__ void combine_kernel(const __bf16* __restrict__ comb,
                               const __bf16* __restrict__ eoT,
                               float* __restrict__ out) {
  const int lane = threadIdx.x & 31;
  const int wave = blockIdx.x * (blockDim.x >> 5) + (threadIdx.x >> 5);
  int mt = wave % (N_ / 16);
  int bh = wave / (N_ / 16);
  int b = bh / H_, h = bh % H_;
  const int r16 = lane & 15, hi = lane >> 4;
  const __bf16* arow = comb + ((size_t)bh * N_ + mt * 16 + r16) * E_;
  v8f acc[4];
#pragma unroll
  for (int dt = 0; dt < 4; ++dt) acc[dt] = (v8f){};
#pragma unroll
  for (int kt = 0; kt < E_; kt += 32) {
    v16bf a = load_a(arow, kt, hi);
#pragma unroll
    for (int dt = 0; dt < 4; ++dt) {
      const __bf16* brow = eoT + ((size_t)b * DIM_ + h * HD_ + dt * 16 + r16) * E_;
      v16bf bm = *(const v16bf*)(brow + kt + hi * 16);
      acc[dt] = wmma_bf16(a, bm, acc[dt]);
    }
  }
#pragma unroll
  for (int dt = 0; dt < 4; ++dt)
#pragma unroll
    for (int r = 0; r < 8; ++r) {
      int n = mt * 16 + hi * 8 + r;
      int d = h * HD_ + dt * 16 + r16;
      out[((size_t)b * N_ + n) * DIM_ + d] = acc[dt][r];
    }
}

// ---------------------------------------------------------------------------
extern "C" void kernel_launch(void* const* d_in, const int* in_sizes, int n_in,
                              void* d_out, int out_size, void* d_ws, size_t ws_size,
                              hipStream_t stream) {
  const float* x   = (const float*)d_in[0];
  const float* phi = (const float*)d_in[1];
  const float* kg  = (const float*)d_in[2];
  const float* kb  = (const float*)d_in[3];
  const float* qg  = (const float*)d_in[4];
  const float* qb  = (const float*)d_in[5];
  const float* lw  = (const float*)d_in[6];
  const float* lb  = (const float*)d_in[7];
  const float* s0  = (const float*)d_in[8];
  const float* s1  = (const float*)d_in[9];
  const float* cw1 = (const float*)d_in[10];
  const float* cb1 = (const float*)d_in[11];
  const float* cw2 = (const float*)d_in[12];
  const float* cb2 = (const float*)d_in[13];
  const float* ow1 = (const float*)d_in[14];
  const float* ob1 = (const float*)d_in[15];
  const float* ow2 = (const float*)d_in[16];
  const float* ob2 = (const float*)d_in[17];
  float* out = (float*)d_out;

  char* ws = (char*)d_ws;
  size_t off = 0;
  auto carve = [&](size_t bytes) -> void* {
    void* p = ws + off;
    off += (bytes + 255) & ~size_t(255);
    return p;
  };
  __bf16* qn_bf    = (__bf16*)carve((size_t)H_ * E_ * HD_ * 2);
  __bf16* k_bf     = (__bf16*)carve((size_t)B_ * H_ * N_ * HD_ * 2);
  __bf16* xT       = (__bf16*)carve((size_t)B_ * H_ * HD_ * N_ * 2);
  float*  logits   = (float*)carve((size_t)B_ * H_ * N_ * E_ * 4);
  __bf16* dispT    = (__bf16*)carve((size_t)B_ * H_ * E_ * N_ * 2);
  __bf16* comb     = (__bf16*)carve((size_t)B_ * H_ * N_ * E_ * 2);
  float*  slots    = (float*)carve((size_t)B_ * E_ * DIM_ * 4);
  float*  hid_core = (float*)carve((size_t)B_ * NC_ * CH_ * 4);
  float*  hid_occ  = (float*)carve((size_t)B_ * NU_ * OH_ * 4);
  __bf16* eoT      = (__bf16*)carve((size_t)B_ * DIM_ * E_ * 2);
  (void)ws_size; (void)in_sizes; (void)n_in; (void)out_size;

  prep_q_kernel<<<H_ * E_, 64, 0, stream>>>(phi, qg, qb, lw, lb, qn_bf);
  prep_k_kernel<<<B_ * N_, DIM_, 0, stream>>>(x, kg, kb, k_bf);
  transpose_x_kernel<<<B_ * H_ * (N_ / 32) * 2, 256, 0, stream>>>(x, xT);

  // logits GEMM: 48*(4096/16) token-stripe waves, 8 waves/block
  logits_kernel<<<(B_ * H_ * (N_ / 16)) / 8, 256, 0, stream>>>(k_bf, qn_bf, logits);

  dispatch_kernel<<<B_ * H_, E_, 0, stream>>>(logits, s0, dispT);
  entmax_kernel<<<B_ * H_ * N_, 256, 0, stream>>>(logits, s1, comb);

  // slots GEMM: 48*10 expert-tile waves
  slots_kernel<<<(B_ * H_ * (E_ / 16)) / 8, 256, 0, stream>>>(dispT, xT, slots);

  mlp1_kernel<<<NC_ * (CH_ / 256), 256, 0, stream>>>(slots, cw1, cb1, hid_core, 0, NC_, CH_, CH_ / 256);
  mlp1_kernel<<<NU_ * (OH_ / 256), 256, 0, stream>>>(slots, ow1, ob1, hid_occ, NC_, NU_, OH_, OH_ / 256);
  mlp2_kernel<CH_><<<NC_, 256, 0, stream>>>(hid_core, cw2, cb2, eoT, 0, NC_);
  mlp2_kernel<OH_><<<NU_, 256, 0, stream>>>(hid_occ, ow2, ob2, eoT, NC_, NU_);

  // combine GEMM: 48*(4096/16) token-stripe waves
  combine_kernel<<<(B_ * H_ * (N_ / 16)) / 8, 256, 0, stream>>>(comb, eoT, out);
}
